// AltBlock_74577812128014
// MI455X (gfx1250) — matrix operand order repaired and checked by reading, not compiled, not run on hardware
//
#include <hip/hip_runtime.h>
#include <math.h>

// ---------------- types / helpers ----------------
typedef __attribute__((ext_vector_type(16))) __bf16 v16bf;
typedef __attribute__((ext_vector_type(8)))  float  v8f;
typedef __attribute__((ext_vector_type(4)))  unsigned int u32x4;
typedef __attribute__((ext_vector_type(8)))  unsigned int u32x8;

union FragU { u32x4 u[2]; v16bf v; };

__device__ __forceinline__ __bf16 f2bf(float x) {
  union { float f; unsigned u; } a; a.f = x;
  unsigned r = a.u + 0x7FFFu + ((a.u >> 16) & 1u);   // round-to-nearest-even
  union { unsigned short s; __bf16 b; } o; o.s = (unsigned short)(r >> 16);
  return o.b;
}
__device__ __forceinline__ v8f vzero8() {
  v8f z;
#pragma unroll
  for (int i = 0; i < 8; ++i) z[i] = 0.f;
  return z;
}

static constexpr int Bb = 2, Nq = 1024, Cc = 1024, Hh = 16, Dd = 64, HIDDEN = 4096;
static constexpr int Mrows = Bb * Nq; // 2048

// ---------------- tiled weight convert + transpose: out[n][k] = bf16(in[k][n]) ----------------
__global__ __launch_bounds__(256) void transpose_to_bf16(const float* __restrict__ in,
                                                         __bf16* __restrict__ out,
                                                         int K, int Ncols) {
  __shared__ __bf16 tile[64 * 66];
  int tx = threadIdx.x & 63, ty = threadIdx.x >> 6;   // ty: 0..3
  int ntn = Ncols >> 6;
  int kt = blockIdx.x / ntn, nt = blockIdx.x % ntn;
  int k0 = kt * 64, n0 = nt * 64;
#pragma unroll
  for (int r = 0; r < 16; ++r) {
    int k = ty * 16 + r;
    tile[tx * 66 + k] = f2bf(in[(size_t)(k0 + k) * Ncols + n0 + tx]); // coalesced read over tx
  }
  __syncthreads();
#pragma unroll
  for (int r = 0; r < 16; ++r) {
    int n = ty * 16 + r;
    out[(size_t)(n0 + n) * K + k0 + tx] = tile[n * 66 + tx];          // coalesced write over tx
  }
}

// ---------------- LayerNorm (row of 1024) -> bf16 ----------------
__global__ __launch_bounds__(256) void layernorm_bf16(const float* __restrict__ x,
                                                      const float* __restrict__ g,
                                                      const float* __restrict__ b,
                                                      __bf16* __restrict__ out) {
  __shared__ float sh[256], sh2[256];
  int row = blockIdx.x, tid = threadIdx.x;
  const float* xr = x + (size_t)row * Cc;
  float s = 0.f, ss = 0.f;
  for (int i = tid; i < Cc; i += 256) { float v = xr[i]; s += v; ss += v * v; }
  sh[tid] = s; sh2[tid] = ss; __syncthreads();
  for (int o = 128; o > 0; o >>= 1) {
    if (tid < o) { sh[tid] += sh[tid + o]; sh2[tid] += sh2[tid + o]; }
    __syncthreads();
  }
  float mean = sh[0] * (1.f / Cc);
  float var  = sh2[0] * (1.f / Cc) - mean * mean;
  float inv  = rsqrtf(var + 1e-5f);
  for (int i = tid; i < Cc; i += 256)
    out[(size_t)row * Cc + i] = f2bf((xr[i] - mean) * inv * g[i] + b[i]);
}

// ---------------- TDM stage of one B panel k-step: 64 cols x 32 ks (bf16, 4KB) ----------------
// 2-D Tensor DMA Descriptor per CDNA5 ISA.
__device__ __forceinline__ void tdm_stage_b(const __bf16* gsrc, unsigned lds_byte_off, int kdim) {
  unsigned long long ga = (unsigned long long)(const void*)gsrc;
  u32x4 g0;
  g0[0] = 1u;                                   // count=1, user descriptor
  g0[1] = lds_byte_off;                         // lds_addr
  g0[2] = (unsigned)ga;                         // global_addr[31:0]
  g0[3] = (unsigned)((ga >> 32) & 0x01FFFFFFu) | 0x80000000u; // addr[56:32] | type=2
  u32x8 g1;
  g1[0] = 0x00010000u;                          // workgroup_mask=0, data_size=1 (2 bytes)
  g1[1] = ((unsigned)kdim & 0xFFFFu) << 16;     // tensor_dim0[15:0] in [31:16]
  g1[2] = ((unsigned)kdim >> 16) | (64u << 16); // tensor_dim0[31:16] | tensor_dim1[15:0]=64
  g1[3] = (32u << 16);                          // tensor_dim1[31:16]=0 | tile_dim0=32
  g1[4] = 64u;                                  // tile_dim1=64, tile_dim2=0
  g1[5] = (unsigned)kdim;                       // tensor_dim0_stride[31:0]
  g1[6] = 0u;                                   // stride[47:32]=0, dim1_stride[15:0]=0
  g1[7] = 0u;
  asm volatile("tensor_load_to_lds %0, %1" :: "s"(g0), "s"(g1) : "memory");
}

// ---------------- generic WMMA GEMM, block tile 128(M) x 64(N), B staged via TDM ----------------
enum { EPI_STORE_F32 = 0, EPI_RESID_F32 = 1, EPI_GELU_BF16 = 2 };

template <int KDIM, int NDIM, int EPI>
__global__ __launch_bounds__(256) void gemm_bf16_wmma(const __bf16* __restrict__ A,
                                                      const __bf16* __restrict__ Bt,
                                                      const float* __restrict__ bias,
                                                      const float* __restrict__ resid,
                                                      void* __restrict__ dst) {
  __shared__ __bf16 lb[2][64 * 32];   // double-buffered B panel (4KB each)
  const int tid = threadIdx.x, lane = tid & 31, wid = tid >> 5;
  const int hi = lane >> 4, ln = lane & 15;
  const int nb = blockIdx.x % (NDIM / 64), mb = blockIdx.x / (NDIM / 64);

  const __bf16* gB = Bt + (size_t)(nb * 64) * KDIM;

  // stage k=0 panel into buffer 0 (wave 0 issues the TDM op for the whole block)
  if (wid == 0) tdm_stage_b(gB, (unsigned)(unsigned long long)(void*)&lb[0][0], KDIM);

  const __bf16* Ab = A + (size_t)(mb * 128 + wid * 16 + ln) * KDIM;
  v8f acc[4];
#pragma unroll
  for (int t = 0; t < 4; ++t) acc[t] = vzero8();

  for (int k = 0; k < KDIM; k += 32) {
    const int buf = (k >> 5) & 1;
    if (k + 32 < KDIM) {
      if (wid == 0) {
        tdm_stage_b(gB + (k + 32), (unsigned)(unsigned long long)(void*)&lb[buf ^ 1][0], KDIM);
        __builtin_amdgcn_s_wait_tensorcnt(1);   // previous stage complete; prefetch in flight
      }
    } else {
      if (wid == 0) __builtin_amdgcn_s_wait_tensorcnt(0);
    }
    __syncthreads();                            // staged panel visible to all waves

    FragU af;
    const __bf16* ap = Ab + k + (hi ? 8 : 0);
    af.u[0] = *(const u32x4*)ap;
    af.u[1] = *(const u32x4*)(ap + 16);
#pragma unroll
    for (int t = 0; t < 4; ++t) {
      const __bf16* bp = &lb[buf][(t * 16 + ln) * 32 + (hi ? 8 : 0)];
      FragU bf;
      bf.u[0] = *(const u32x4*)bp;
      bf.u[1] = *(const u32x4*)(bp + 16);
      acc[t] = __builtin_amdgcn_wmma_f32_16x16x32_bf16(false, af.v, false, bf.v,
                                                       (short)0, acc[t], false, false);
    }
    __syncthreads();                            // all reads done before panel is overwritten
  }

#pragma unroll
  for (int t = 0; t < 4; ++t) {
    int nc = nb * 64 + t * 16 + ln;
    float bs = bias[nc];
#pragma unroll
    for (int r = 0; r < 8; ++r) {
      int m = mb * 128 + wid * 16 + r + 8 * hi;
      size_t oi = (size_t)m * NDIM + nc;
      float v = acc[t][r] + bs;
      if constexpr (EPI == EPI_STORE_F32) {
        ((float*)dst)[oi] = v;
      } else if constexpr (EPI == EPI_RESID_F32) {
        ((float*)dst)[oi] = v + resid[oi];
      } else {
        ((__bf16*)dst)[oi] = f2bf(0.5f * v * (1.0f + erff(v * 0.70710678118654752f)));
      }
    }
  }
}

// ---------------- QKV post: normalize q,k -> bf16 [B,H,N,D]; v -> bf16 [B,H,D,N] ----------------
__global__ __launch_bounds__(256) void qkv_post(const float* __restrict__ qkv,
                                                __bf16* __restrict__ qn,
                                                __bf16* __restrict__ kn,
                                                __bf16* __restrict__ vt) {
  int lane = threadIdx.x & 31, wid = threadIdx.x >> 5;
  int idx = blockIdx.x * 8 + wid;          // (b,h,n) flattened: bh*1024 + n
  int n = idx & (Nq - 1);
  int bh = idx >> 10;
  int h = bh & (Hh - 1), b = bh >> 4;
  size_t row = (size_t)(b * Nq + n) * (3 * Cc);
  int col = h * Dd;
  float q0 = qkv[row + col + lane],            q1 = qkv[row + col + lane + 32];
  float k0 = qkv[row + Cc + col + lane],       k1 = qkv[row + Cc + col + lane + 32];
  float v0 = qkv[row + 2 * Cc + col + lane],   v1 = qkv[row + 2 * Cc + col + lane + 32];
  float qs = q0 * q0 + q1 * q1, ks = k0 * k0 + k1 * k1;
#pragma unroll
  for (int o = 16; o >= 1; o >>= 1) {
    qs += __shfl_xor(qs, o, 32);
    ks += __shfl_xor(ks, o, 32);
  }
  float qi = 1.f / fmaxf(sqrtf(qs), 1e-12f);
  float ki = 1.f / fmaxf(sqrtf(ks), 1e-12f);
  size_t o = ((size_t)bh * Nq + n) * Dd;
  qn[o + lane] = f2bf(q0 * qi); qn[o + lane + 32] = f2bf(q1 * qi);
  kn[o + lane] = f2bf(k0 * ki); kn[o + lane + 32] = f2bf(k1 * ki);
  size_t vo = (size_t)bh * Dd * Nq;
  vt[vo + (size_t)lane * Nq + n]        = f2bf(v0);
  vt[vo + (size_t)(lane + 32) * Nq + n] = f2bf(v1);
}

// ---------------- flash attention: one wave = 16 query rows of one (b,h) ----------------
__global__ __launch_bounds__(256) void attn_kernel(const __bf16* __restrict__ qn,
                                                   const __bf16* __restrict__ kn,
                                                   const __bf16* __restrict__ vt,
                                                   const float* __restrict__ alibi,
                                                   const unsigned char* __restrict__ mask,
                                                   const float* __restrict__ logit_scale,
                                                   __bf16* __restrict__ out /* [B,N,C] bf16 */) {
  __shared__ __bf16 lds_p[8][16 * 32];
  int lane = threadIdx.x & 31, wid = threadIdx.x >> 5;
  int sid = blockIdx.x * 8 + wid;          // 0 .. B*H*64-1
  int mt = sid & 63;                        // query tile
  int bh = sid >> 6;
  int h = bh & (Hh - 1), b = bh >> 4;
  int hi = lane >> 4, ln = lane & 15;

  float scale = __expf(fminf(logit_scale[h], 4.605170185988091f)); // log(100)

  const __bf16* qb = qn + (size_t)bh * Nq * Dd;
  const __bf16* kb = kn + (size_t)bh * Nq * Dd;
  const __bf16* vb = vt + (size_t)bh * Dd * Nq;
  const float* ab = alibi + (size_t)bh * Nq * Nq;
  const unsigned char* mb = mask + b * Nq;

  // Q fragments (D=64 -> two K-steps of 32)
  FragU qa[2];
  {
    const __bf16* p = qb + (size_t)(mt * 16 + ln) * Dd + (hi ? 8 : 0);
#pragma unroll
    for (int kk = 0; kk < 2; ++kk) {
      qa[kk].u[0] = *(const u32x4*)(p + kk * 32);
      qa[kk].u[1] = *(const u32x4*)(p + kk * 32 + 16);
    }
  }

  v8f o0[4];
#pragma unroll
  for (int t = 0; t < 4; ++t) o0[t] = vzero8();
  float mrow[8], lrow[8];
#pragma unroll
  for (int r = 0; r < 8; ++r) { mrow[r] = -INFINITY; lrow[r] = 0.f; }

  for (int j = 0; j < Nq; j += 32) {
    // keep the HBM-dominant alibi stream ahead: prefetch next tile's rows
    if (hi == 0 && j + 32 < Nq)
      __builtin_prefetch(ab + (size_t)(mt * 16 + ln) * Nq + j + 32, 0, 3);

    v8f s[2]; s[0] = vzero8(); s[1] = vzero8();
#pragma unroll
    for (int sub = 0; sub < 2; ++sub) {
      int key = j + sub * 16 + ln;
#pragma unroll
      for (int kk = 0; kk < 2; ++kk) {
        FragU bf;
        const __bf16* p = kb + (size_t)key * Dd + kk * 32 + (hi ? 8 : 0);
        bf.u[0] = *(const u32x4*)p;
        bf.u[1] = *(const u32x4*)(p + 16);
        s[sub] = __builtin_amdgcn_wmma_f32_16x16x32_bf16(false, qa[kk].v, false, bf.v,
                                                         (short)0, s[sub], false, false);
      }
      float madd = mb[key] ? -INFINITY : 0.f;
#pragma unroll
      for (int r = 0; r < 8; ++r) {
        int m = mt * 16 + r + 8 * hi;
        s[sub][r] = s[sub][r] * scale + ab[(size_t)m * Nq + key] + madd;
      }
    }
    // online softmax per row (rows live as (r, lane-half); cols across 16 lanes)
#pragma unroll
    for (int r = 0; r < 8; ++r) {
      float mx = fmaxf(s[0][r], s[1][r]);
#pragma unroll
      for (int o = 8; o >= 1; o >>= 1) mx = fmaxf(mx, __shfl_xor(mx, o, 32));
      float mnew = fmaxf(mrow[r], mx);
      float alpha = (mrow[r] == -INFINITY) ? 0.f : __expf(mrow[r] - mnew);
      float p0, p1;
      if (mnew == -INFINITY) { p0 = 0.f; p1 = 0.f; alpha = 0.f; }
      else { p0 = __expf(s[0][r] - mnew); p1 = __expf(s[1][r] - mnew); }
      float rs = p0 + p1;
#pragma unroll
      for (int o = 8; o >= 1; o >>= 1) rs += __shfl_xor(rs, o, 32);
      lrow[r] = lrow[r] * alpha + rs;
      mrow[r] = mnew;
#pragma unroll
      for (int t = 0; t < 4; ++t) o0[t][r] *= alpha;
      int m = r + 8 * hi;
      lds_p[wid][m * 32 + ln]      = f2bf(p0);
      lds_p[wid][m * 32 + 16 + ln] = f2bf(p1);
    }
    asm volatile("s_wait_dscnt 0" ::: "memory");
    FragU pa;
    {
      const __bf16* p = &lds_p[wid][ln * 32 + (hi ? 8 : 0)];
      pa.u[0] = *(const u32x4*)p;
      pa.u[1] = *(const u32x4*)(p + 16);
    }
#pragma unroll
    for (int t = 0; t < 4; ++t) {
      FragU bf;
      const __bf16* p = vb + (size_t)(t * 16 + ln) * Nq + j + (hi ? 8 : 0);
      bf.u[0] = *(const u32x4*)p;
      bf.u[1] = *(const u32x4*)(p + 16);
      o0[t] = __builtin_amdgcn_wmma_f32_16x16x32_bf16(false, pa.v, false, bf.v,
                                                      (short)0, o0[t], false, false);
    }
  }
#pragma unroll
  for (int t = 0; t < 4; ++t)
#pragma unroll
    for (int r = 0; r < 8; ++r) {
      int m = mt * 16 + r + 8 * hi;
      float inv = (lrow[r] > 0.f) ? 1.f / lrow[r] : 0.f;
      out[((size_t)b * Nq + m) * Cc + h * Dd + t * 16 + ln] = f2bf(o0[t][r] * inv);
    }
}

// ---------------- host side ----------------
extern "C" void kernel_launch(void* const* d_in, const int* in_sizes, int n_in,
                              void* d_out, int out_size, void* d_ws, size_t ws_size,
                              hipStream_t stream) {
  const float* x      = (const float*)d_in[0];
  const unsigned char* mask = (const unsigned char*)d_in[1];
  const float* alibi  = (const float*)d_in[2];
  const float* ln1g   = (const float*)d_in[3];
  const float* ln1b   = (const float*)d_in[4];
  const float* Wqkv   = (const float*)d_in[5];
  const float* bqkv   = (const float*)d_in[6];
  const float* lscale = (const float*)d_in[7];
  const float* Wproj  = (const float*)d_in[8];
  const float* bproj  = (const float*)d_in[9];
  const float* ln2g   = (const float*)d_in[10];
  const float* ln2b   = (const float*)d_in[11];
  const float* W1     = (const float*)d_in[12];
  const float* b1     = (const float*)d_in[13];
  const float* W2     = (const float*)d_in[14];
  const float* b2     = (const float*)d_in[15];
  float* out = (float*)d_out;

  char* ws = (char*)d_ws;
  auto alloc = [&](size_t bytes) -> char* {
    char* p = ws; ws += (bytes + 255) & ~(size_t)255; return p;
  };
  __bf16* wqkv_t = (__bf16*)alloc((size_t)3 * Cc * Cc * 2);        // [3072][1024]
  __bf16* wproj_t = (__bf16*)alloc((size_t)Cc * Cc * 2);           // [1024][1024]
  __bf16* w1_t   = (__bf16*)alloc((size_t)HIDDEN * Cc * 2);        // [4096][1024]
  __bf16* w2_t   = (__bf16*)alloc((size_t)Cc * HIDDEN * 2);        // [1024][4096]
  __bf16* h_bf   = (__bf16*)alloc((size_t)Mrows * Cc * 2);
  float*  qkv_f  = (float*)alloc((size_t)Mrows * 3 * Cc * 4);
  __bf16* qn     = (__bf16*)alloc((size_t)Bb * Hh * Nq * Dd * 2);
  __bf16* kn     = (__bf16*)alloc((size_t)Bb * Hh * Nq * Dd * 2);
  __bf16* vt     = (__bf16*)alloc((size_t)Bb * Hh * Dd * Nq * 2);
  __bf16* att_bf = (__bf16*)alloc((size_t)Mrows * Cc * 2);
  float*  x1     = (float*)alloc((size_t)Mrows * Cc * 4);
  __bf16* y_bf   = (__bf16*)alloc((size_t)Mrows * Cc * 2);
  __bf16* hid_bf = (__bf16*)alloc((size_t)Mrows * HIDDEN * 2);

  // weight convert+transpose (tiled, coalesced both sides)
  transpose_to_bf16<<<(Cc / 64) * (3 * Cc / 64), 256, 0, stream>>>(Wqkv, wqkv_t, Cc, 3 * Cc);
  transpose_to_bf16<<<(Cc / 64) * (Cc / 64), 256, 0, stream>>>(Wproj, wproj_t, Cc, Cc);
  transpose_to_bf16<<<(Cc / 64) * (HIDDEN / 64), 256, 0, stream>>>(W1, w1_t, Cc, HIDDEN);
  transpose_to_bf16<<<(HIDDEN / 64) * (Cc / 64), 256, 0, stream>>>(W2, w2_t, HIDDEN, Cc);

  // LN1 -> h_bf
  layernorm_bf16<<<Mrows, 256, 0, stream>>>(x, ln1g, ln1b, h_bf);

  // QKV GEMM: [2048x1024] x [1024x3072] + bqkv -> f32
  gemm_bf16_wmma<Cc, 3 * Cc, EPI_STORE_F32>
      <<<(Mrows / 128) * (3 * Cc / 64), 256, 0, stream>>>(h_bf, wqkv_t, bqkv, nullptr, qkv_f);

  // normalize q,k; transpose v
  qkv_post<<<(Bb * Hh * Nq) / 8, 256, 0, stream>>>(qkv_f, qn, kn, vt);

  // attention
  attn_kernel<<<(Bb * Hh * (Nq / 16)) / 8, 256, 0, stream>>>(qn, kn, vt, alibi, mask, lscale, att_bf);

  // proj GEMM + residual: x1 = x + att@Wproj + bproj
  gemm_bf16_wmma<Cc, Cc, EPI_RESID_F32>
      <<<(Mrows / 128) * (Cc / 64), 256, 0, stream>>>(att_bf, wproj_t, bproj, x, x1);

  // LN2 -> y_bf
  layernorm_bf16<<<Mrows, 256, 0, stream>>>(x1, ln2g, ln2b, y_bf);

  // MLP1 + GELU -> bf16
  gemm_bf16_wmma<Cc, HIDDEN, EPI_GELU_BF16>
      <<<(Mrows / 128) * (HIDDEN / 64), 256, 0, stream>>>(y_bf, w1_t, b1, nullptr, hid_bf);

  // MLP2 + residual -> out
  gemm_bf16_wmma<HIDDEN, Cc, EPI_RESID_F32>
      <<<(Mrows / 128) * (Cc / 64), 256, 0, stream>>>(hid_bf, w2_t, b2, x1, out);
}
